// DiT_29231547416683
// MI455X (gfx1250) — matrix operand order repaired and checked by reading, not compiled, not run on hardware
//
#include <hip/hip_runtime.h>

// ---------------- problem constants ----------------
#define LAYERS 8
#define BATCH  8
#define NTOK   512
#define CDIM   1024
#define HEADS  16
#define HDIM   64
#define MROWS  (BATCH*NTOK)   // 4096
#define INV_TP 0.08838834764831845f  // 1/sqrt(2*64)

typedef __bf16 bf16_t;
typedef __attribute__((ext_vector_type(16))) __bf16 v16bf;
typedef __attribute__((ext_vector_type(8)))  __bf16 v8bf;
typedef __attribute__((ext_vector_type(8)))  float  v8f;

static __device__ __forceinline__ v16bf cat8(v8bf lo, v8bf hi) {
  return __builtin_shufflevector(lo, hi, 0,1,2,3,4,5,6,7,8,9,10,11,12,13,14,15);
}
static __device__ __forceinline__ float seluf(float x) {
  const float sc = 1.0507009873554805f, al = 1.6732632423543772f;
  return x > 0.f ? sc * x : sc * al * (__expf(x) - 1.f);
}

// ---------------- Tensor Data Mover helpers (CDNA5) ----------------
#if __has_builtin(__builtin_amdgcn_tensor_load_to_lds)
#define HAVE_TDM 1
typedef __attribute__((ext_vector_type(4))) unsigned int u32x4_t;
typedef __attribute__((ext_vector_type(8))) int i32x8_t;
typedef __attribute__((ext_vector_type(4))) int i32x4_t;

// 2D tile DMA: tile_x elements (2B each) * tile_rows rows, row stride in elements.
// LDS rows padded via TDM pad feature: pad_int_code (interval, dwords=2<<code),
// pad_amt_code (amount, dwords=code+1).
static __device__ __forceinline__ void tdm_load_tile_2d(
    const void* gsrc, unsigned lds_off, unsigned tile_x, unsigned tile_rows,
    unsigned row_stride_elems, unsigned pad_int_code, unsigned pad_amt_code) {
  unsigned long long ga = (unsigned long long)(uintptr_t)gsrc;
  u32x4_t g0;
  g0[0] = 1u;                                                    // count=1, user mode
  g0[1] = lds_off;                                               // lds_addr
  g0[2] = (unsigned)ga;                                          // global_addr[31:0]
  g0[3] = (unsigned)((ga >> 32) & 0x01ffffffu) | (2u << 30);     // addr[56:32] | type=2
  i32x8_t g1;
  unsigned dw0 = (1u << 16)                                      // data_size = 2 bytes
               | (1u << 20)                                      // pad_enable
               | (pad_int_code << 22) | (pad_amt_code << 25);
  unsigned td0 = row_stride_elems;                               // tensor_dim0
  unsigned td1 = 0x7fffffffu;                                    // tensor_dim1 (large)
  g1[0] = (int)dw0;
  g1[1] = (int)((td0 & 0xffffu) << 16);                          // tensor_dim0[15:0] @bit48
  g1[2] = (int)(((td0 >> 16) & 0xffffu) | ((td1 & 0xffffu) << 16));
  g1[3] = (int)(((td1 >> 16) & 0xffffu) | ((tile_x & 0xffffu) << 16)); // tile_dim0
  g1[4] = (int)(tile_rows & 0xffffu);                            // tile_dim1 (tile_dim2=0)
  g1[5] = (int)row_stride_elems;                                 // tensor_dim0_stride lo32
  g1[6] = 0;                                                     // stride hi / dim1_stride lo
  g1[7] = 0;
  i32x4_t z4 = {0, 0, 0, 0};
  i32x8_t z8 = {0, 0, 0, 0, 0, 0, 0, 0};
  // 6-arg form (clang-23 / therock-10.0 headers): groups 0..3 + aux + cpol
  __builtin_amdgcn_tensor_load_to_lds(g0, g1, z4, z4, z8, 0);
}
static __device__ __forceinline__ void tdm_wait0() {
#if __has_builtin(__builtin_amdgcn_s_wait_tensorcnt)
  __builtin_amdgcn_s_wait_tensorcnt(0);
#else
  asm volatile("s_wait_tensorcnt 0x0" ::: "memory");
#endif
}
static __device__ __forceinline__ unsigned lds_off_of(const void* p) {
  return (unsigned)(uintptr_t)p;   // LDS aperture: low 32 bits = LDS byte offset
}
#else
#define HAVE_TDM 0
#endif

// ---------------- fp32 -> bf16 convert ----------------
__global__ void f2bf_kernel(const float* __restrict__ in, bf16_t* __restrict__ out, int n) {
  int i = blockIdx.x * 256 + threadIdx.x;
  if (i < n) out[i] = (bf16_t)in[i];
}

// ---------------- conditioning embed: seluc = selu(lemb + te) ----------------
__global__ void cond_embed_kernel(const float* __restrict__ time_, const float* __restrict__ label,
                                  const float* __restrict__ Wl, const float* __restrict__ bl,
                                  float* __restrict__ seluc) {
  int i = blockIdx.x * 256 + threadIdx.x;             // B*C = 8192
  if (i >= BATCH * CDIM) return;
  int b = i >> 10, ch = i & (CDIM - 1);
  float lemb = label[b*2+0]*Wl[ch*2+0] + label[b*2+1]*Wl[ch*2+1] + bl[ch];
  float base = __powf(10000.f, 2.f * (float)ch / (float)CDIM);
  float tt   = time_[b] / base;
  float te   = ((ch & 1) == 0) ? __sinf(tt) : __cosf(tt);
  seluc[i] = seluf(lemb + te);
}

// ---------------- tiny conditioning linear ----------------
__global__ __launch_bounds__(256)
void cond_linear_kernel(const float* __restrict__ seluc, const float* __restrict__ W,
                        const float* __restrict__ bias, float* __restrict__ out, int Nout) {
  int j = blockIdx.x;
  int t = threadIdx.x;
  const float* wr = W + (size_t)j * CDIM;
  float part[BATCH];
#pragma unroll
  for (int b = 0; b < BATCH; b++) part[b] = 0.f;
  for (int k = t; k < CDIM; k += 256) {
    float w = wr[k];
#pragma unroll
    for (int b = 0; b < BATCH; b++) part[b] += w * seluc[b * CDIM + k];
  }
  __shared__ float red[BATCH][256];
#pragma unroll
  for (int b = 0; b < BATCH; b++) red[b][t] = part[b];
  __syncthreads();
  for (int o = 128; o > 0; o >>= 1) {
    if (t < o) {
#pragma unroll
      for (int b = 0; b < BATCH; b++) red[b][t] += red[b][t + o];
    }
    __syncthreads();
  }
  if (t < BATCH) out[(size_t)t * Nout + j] = red[t][0] + bias[j];
}

// ---------------- layernorm + modulate -> bf16 ----------------
__global__ __launch_bounds__(256)
void ln_mod_kernel(const float* __restrict__ xc, const float* __restrict__ mod, int mstride,
                   int shift_off, int scale_off, const float* __restrict__ xmask, int use_mask,
                   bf16_t* __restrict__ outH) {
  int m = blockIdx.x;
  int t = threadIdx.x;
  int b = m >> 9;
  const float* xr = xc + (size_t)m * CDIM;
  float v[4], s = 0.f, s2 = 0.f;
#pragma unroll
  for (int i = 0; i < 4; i++) {
    v[i] = xr[t + i * 256];
    s += v[i]; s2 += v[i] * v[i];
  }
  __shared__ float r1[256], r2[256];
  r1[t] = s; r2[t] = s2;
  __syncthreads();
  for (int o = 128; o > 0; o >>= 1) {
    if (t < o) { r1[t] += r1[t + o]; r2[t] += r2[t + o]; }
    __syncthreads();
  }
  float mean = r1[0] * (1.f / CDIM);
  float var  = r2[0] * (1.f / CDIM) - mean * mean;
  float inv  = rsqrtf(var + 1e-6f);
  float msk  = use_mask ? xmask[m] : 1.f;
#pragma unroll
  for (int i = 0; i < 4; i++) {
    int ch = t + i * 256;
    float sh = mod[(size_t)b * mstride + shift_off + ch];
    float sc = mod[(size_t)b * mstride + scale_off + ch];
    float hv = ((v[i] - mean) * inv) * (1.f + sc) + sh;
    outH[(size_t)m * CDIM + ch] = (bf16_t)(hv * msk);
  }
}

// ---------------- residual gate: xc += gate * att ----------------
__global__ void resid_gate_kernel(float* __restrict__ xc, const float* __restrict__ att,
                                  const float* __restrict__ mod, int mstride, int gate_off) {
  size_t i = (size_t)blockIdx.x * 256 + threadIdx.x;
  int m = (int)(i >> 10), ch = (int)(i & (CDIM - 1)), b = m >> 9;
  xc[i] += mod[(size_t)b * mstride + gate_off + ch] * att[i];
}

// ---------------- bf16 WMMA GEMM with TDM-fed LDS tiles ----------------
// mode 0: outH = acc (bf16)
// mode 1: outH = selu(acc + bias) (bf16)
// mode 2: resid = (resid + gate*(acc + bias)) * xmask
// mode 3: outF = (acc + bias) * xmask
#define GK 32
#define LKP 40   // padded K stride in halfs (80B rows; TDM pad: 16dw interval + 4dw pad)
__global__ __launch_bounds__(256)
void gemm_bf16_wmma_kernel(const bf16_t* __restrict__ A, const bf16_t* __restrict__ Wt,
                           const float* __restrict__ bias,
                           float* __restrict__ outF, bf16_t* __restrict__ outH,
                           float* __restrict__ resid, const float* __restrict__ gate, int gstride,
                           const float* __restrict__ xmask,
                           int M, int Nn, int K, int mode) {
  __shared__ __align__(16) bf16_t As[2][128][LKP];
  __shared__ __align__(16) bf16_t Bs[2][128][LKP];

  const int t    = threadIdx.x;
  const int lane = t & 31;
  const int wave = t >> 5;
  const int wm   = wave >> 2;          // 0..1  (64-row strip)
  const int wn   = wave & 3;           // 0..3  (32-col strip)
  const int lr   = lane & 15;
  const int lh   = lane >> 4;
  const int m0   = blockIdx.y * 128;
  const int n0   = blockIdx.x * 128;

#if !HAVE_TDM
  auto load_tile = [&](int buf, int kk) {
#pragma unroll
    for (int p = 0; p < 2; p++) {
      int idx = p * 256 + t;
      int r   = idx >> 2;
      int c8  = (idx & 3) * 8;
      *(uint4*)&As[buf][r][c8] = *(const uint4*)(A  + (size_t)(m0 + r) * K + kk + c8);
      *(uint4*)&Bs[buf][r][c8] = *(const uint4*)(Wt + (size_t)(n0 + r) * K + kk + c8);
    }
  };
#endif

  const v8f vzero = {0.f,0.f,0.f,0.f,0.f,0.f,0.f,0.f};
  v8f acc[4][2];
#pragma unroll
  for (int mi = 0; mi < 4; mi++)
#pragma unroll
    for (int ni = 0; ni < 2; ni++) acc[mi][ni] = vzero;

  const int nk = K / GK;

#if HAVE_TDM
  if (wave == 0) {
    tdm_load_tile_2d(A  + (size_t)m0 * K, lds_off_of(&As[0][0][0]), GK, 128, K, 3, 3);
    tdm_load_tile_2d(Wt + (size_t)n0 * K, lds_off_of(&Bs[0][0][0]), GK, 128, K, 3, 3);
  }
#else
  load_tile(0, 0);
#endif

  for (int ks = 0; ks < nk; ks++) {
    int buf = ks & 1;
#if HAVE_TDM
    if (wave == 0) tdm_wait0();          // current buf's 2 DMA ops complete
    __syncthreads();                     // publish to all waves
    if (ks + 1 < nk && wave == 0) {      // overlap next tile DMA with WMMA work
      tdm_load_tile_2d(A  + (size_t)m0 * K + (ks + 1) * GK,
                       lds_off_of(&As[buf ^ 1][0][0]), GK, 128, K, 3, 3);
      tdm_load_tile_2d(Wt + (size_t)n0 * K + (ks + 1) * GK,
                       lds_off_of(&Bs[buf ^ 1][0][0]), GK, 128, K, 3, 3);
    }
#else
    __syncthreads();
    if (ks + 1 < nk) load_tile(buf ^ 1, (ks + 1) * GK);
    if (ks + 3 < nk) {
      __builtin_prefetch(A  + (size_t)(m0 + (t & 127)) * K + (ks + 3) * GK, 0, 1);
      __builtin_prefetch(Wt + (size_t)(n0 + (t & 127)) * K + (ks + 3) * GK, 0, 1);
    }
#endif

    v16bf af[4], bf2[2];
#pragma unroll
    for (int mi = 0; mi < 4; mi++) {
      const bf16_t* ap = &As[buf][wm * 64 + mi * 16 + lr][0];
      v8bf lo = *(const v8bf*)(ap + lh * 8);
      v8bf hi = *(const v8bf*)(ap + 16 + lh * 8);
      af[mi] = cat8(lo, hi);
    }
#pragma unroll
    for (int ni = 0; ni < 2; ni++) {
      const bf16_t* bp = &Bs[buf][wn * 32 + ni * 16 + lr][0];
      v8bf lo = *(const v8bf*)(bp + lh * 8);
      v8bf hi = *(const v8bf*)(bp + 16 + lh * 8);
      bf2[ni] = cat8(lo, hi);
    }
#pragma unroll
    for (int mi = 0; mi < 4; mi++)
#pragma unroll
      for (int ni = 0; ni < 2; ni++)
        acc[mi][ni] = __builtin_amdgcn_wmma_f32_16x16x32_bf16(
            false, af[mi], false, bf2[ni], (short)0, acc[mi][ni], false, false);
  }

  // epilogue: element e -> M = e + lh*8, N = lr within 16x16 subtile
#pragma unroll
  for (int mi = 0; mi < 4; mi++) {
#pragma unroll
    for (int ni = 0; ni < 2; ni++) {
#pragma unroll
      for (int e = 0; e < 8; e++) {
        int mr = m0 + wm * 64 + mi * 16 + e + lh * 8;
        int nc = n0 + wn * 32 + ni * 16 + lr;
        float v = acc[mi][ni][e];
        size_t o = (size_t)mr * Nn + nc;
        if (mode == 0) {
          outH[o] = (bf16_t)v;
        } else if (mode == 1) {
          outH[o] = (bf16_t)seluf(v + bias[nc]);
        } else if (mode == 2) {
          int b = mr >> 9;
          float g = gate[(size_t)b * gstride + nc];
          resid[o] = (resid[o] + g * (v + bias[nc])) * xmask[mr];
        } else {
          outF[o] = (v + bias[nc]) * xmask[mr];
        }
      }
    }
  }
}

// ---------------- flash attention with WMMA (per (b,h), 64-row q tile) ----------------
__global__ __launch_bounds__(256)
void attn_wmma_kernel(const bf16_t* __restrict__ qkv,  // [4096][3072]: q|k|v
                      const float* __restrict__ xmask, float* __restrict__ att) {
  __shared__ __align__(16) bf16_t Qs[64][72];
  __shared__ __align__(16) bf16_t Ks[64][72];
  __shared__ __align__(16) bf16_t Vts[64][72];   // transposed: [d][k]
  __shared__ __align__(16) bf16_t Ps[64][72];
  __shared__ float Ss[64][68];
  __shared__ float xq[64], xk[64];
  __shared__ float tred[64][4];
  __shared__ float stat_m[64], stat_l[64], rs_s[64], nm_s[64];

  const int t    = threadIdx.x;
  const int lane = t & 31;
  const int wave = t >> 5;
  const int lr   = lane & 15;
  const int lh   = lane >> 4;
  const int bh   = blockIdx.x;
  const int b    = bh >> 4;
  const int h    = bh & 15;
  const int q0   = blockIdx.y * 64;

  const int owm = wave >> 1;   // 0..3 (16-row strip)
  const int own = wave & 1;    // 0..1 (32-col strip)
  const int swm = owm, swn = own;

  const bf16_t* qbase = qkv + ((size_t)(b * NTOK + q0)) * 3072 + h * 64;
#if HAVE_TDM
  if (wave == 0)
    tdm_load_tile_2d(qbase, lds_off_of(&Qs[0][0]), 64, 64, 3072, 4, 3);
#else
#pragma unroll
  for (int p = 0; p < 2; p++) {
    int idx = p * 256 + t;
    int r = idx >> 3, c8 = (idx & 7) * 8;
    *(uint4*)&Qs[r][c8] = *(const uint4*)(qbase + (size_t)r * 3072 + c8);
  }
#endif
  if (t < 64) {
    xq[t] = xmask[b * NTOK + q0 + t];
    stat_m[t] = -1e30f;
    stat_l[t] = 0.f;
  }

  const v8f vzero = {0.f,0.f,0.f,0.f,0.f,0.f,0.f,0.f};
  v8f acc_o[2] = {vzero, vzero};

  for (int kt = 0; kt < NTOK / 64; kt++) {
    int kk = kt * 64;
    const bf16_t* kbase = qkv + ((size_t)(b * NTOK + kk)) * 3072 + 1024 + h * 64;
    const bf16_t* vbase = qkv + ((size_t)(b * NTOK + kk)) * 3072 + 2048 + h * 64;
#if HAVE_TDM
    if (wave == 0)
      tdm_load_tile_2d(kbase, lds_off_of(&Ks[0][0]), 64, 64, 3072, 4, 3);
#else
#pragma unroll
    for (int p = 0; p < 2; p++) {
      int idx = p * 256 + t;
      int r = idx >> 3, c8 = (idx & 7) * 8;
      *(uint4*)&Ks[r][c8] = *(const uint4*)(kbase + (size_t)r * 3072 + c8);
    }
#endif
#pragma unroll
    for (int i = 0; i < 16; i++) {
      int idx = i * 256 + t;               // V transpose (element-wise)
      int r = idx >> 6, d = idx & 63;
      Vts[d][r] = vbase[(size_t)r * 3072 + d];
    }
    if (t < 64) xk[t] = xmask[b * NTOK + kk + t];
#if HAVE_TDM
    if (wave == 0) tdm_wait0();
#endif
    __syncthreads();

    // S = Q K^T / TP  (+mask), written to Ss
#pragma unroll
    for (int ni = 0; ni < 2; ni++) {
      v8f s = vzero;
#pragma unroll
      for (int k2 = 0; k2 < 2; k2++) {
        const bf16_t* ap = &Qs[swm * 16 + lr][k2 * 32];
        v16bf a = cat8(*(const v8bf*)(ap + lh * 8), *(const v8bf*)(ap + 16 + lh * 8));
        const bf16_t* bp = &Ks[swn * 32 + ni * 16 + lr][k2 * 32];
        v16bf bb = cat8(*(const v8bf*)(bp + lh * 8), *(const v8bf*)(bp + 16 + lh * 8));
        s = __builtin_amdgcn_wmma_f32_16x16x32_bf16(false, a, false, bb, (short)0, s, false, false);
      }
#pragma unroll
      for (int e = 0; e < 8; e++) {
        int qr = swm * 16 + e + lh * 8;
        int kc = swn * 32 + ni * 16 + lr;
        float val = s[e] * INV_TP;
        if (xq[qr] + xk[kc] == 1.0f) val = -3.0e38f;
        Ss[qr][kc] = val;
      }
    }
    __syncthreads();

    // online softmax over this 64-wide tile
    int row = t >> 2, seg = t & 3;
    float lm = -3.0e38f;
#pragma unroll
    for (int j = 0; j < 16; j++) lm = fmaxf(lm, Ss[row][seg * 16 + j]);
    tred[row][seg] = lm;
    __syncthreads();
    if (seg == 0) {
      float rm = fmaxf(fmaxf(tred[row][0], tred[row][1]), fmaxf(tred[row][2], tred[row][3]));
      float m_old = stat_m[row];
      float m_new = fmaxf(m_old, rm);
      nm_s[row] = m_new;
      rs_s[row] = __expf(m_old - m_new);
    }
    __syncthreads();
    float m_new = nm_s[row];
    float ls = 0.f;
#pragma unroll
    for (int j = 0; j < 16; j++) {
      float p = __expf(Ss[row][seg * 16 + j] - m_new);
      ls += p;
      Ps[row][seg * 16 + j] = (bf16_t)p;
    }
    tred[row][seg] = ls;
    __syncthreads();
    if (seg == 0) {
      stat_l[row] = stat_l[row] * rs_s[row] + tred[row][0] + tred[row][1] + tred[row][2] + tred[row][3];
      stat_m[row] = m_new;
    }
    __syncthreads();

    // O = O*rs + P V
#pragma unroll
    for (int ni = 0; ni < 2; ni++) {
#pragma unroll
      for (int e = 0; e < 8; e++) {
        int qr = owm * 16 + e + lh * 8;
        acc_o[ni][e] *= rs_s[qr];
      }
#pragma unroll
      for (int k2 = 0; k2 < 2; k2++) {
        const bf16_t* ap = &Ps[owm * 16 + lr][k2 * 32];
        v16bf a = cat8(*(const v8bf*)(ap + lh * 8), *(const v8bf*)(ap + 16 + lh * 8));
        const bf16_t* bp = &Vts[own * 32 + ni * 16 + lr][k2 * 32];
        v16bf bb = cat8(*(const v8bf*)(bp + lh * 8), *(const v8bf*)(bp + 16 + lh * 8));
        acc_o[ni] = __builtin_amdgcn_wmma_f32_16x16x32_bf16(false, a, false, bb, (short)0, acc_o[ni], false, false);
      }
    }
    __syncthreads();
  }

  float* obase = att + ((size_t)(b * NTOK + q0)) * CDIM + h * 64;
#pragma unroll
  for (int ni = 0; ni < 2; ni++) {
#pragma unroll
    for (int e = 0; e < 8; e++) {
      int qr = owm * 16 + e + lh * 8;
      int d  = own * 32 + ni * 16 + lr;
      obase[(size_t)qr * CDIM + d] = acc_o[ni][e] / stat_l[qr];
    }
  }
}

// ---------------- host launch ----------------
extern "C" void kernel_launch(void* const* d_in, const int* in_sizes, int n_in,
                              void* d_out, int out_size, void* d_ws, size_t ws_size,
                              hipStream_t stream) {
  const float* x     = (const float*)d_in[0];
  const float* time_ = (const float*)d_in[1];
  const float* label = (const float*)d_in[2];
  const float* xmask = (const float*)d_in[3];
  const float* Wl    = (const float*)d_in[4];
  const float* bl    = (const float*)d_in[5];
  const float* Wq    = (const float*)d_in[6];
  const float* Wk    = (const float*)d_in[7];
  const float* Wv    = (const float*)d_in[8];
  const float* W1    = (const float*)d_in[9];
  const float* b1    = (const float*)d_in[10];
  const float* W2    = (const float*)d_in[11];
  const float* b2    = (const float*)d_in[12];
  const float* Wm    = (const float*)d_in[13];
  const float* bm    = (const float*)d_in[14];
  const float* Wf    = (const float*)d_in[15];
  const float* bf_   = (const float*)d_in[16];
  const float* Wmf   = (const float*)d_in[17];
  const float* bmf   = (const float*)d_in[18];
  float* out = (float*)d_out;

  uint8_t* p = (uint8_t*)d_ws;
  auto alloc = [&](size_t bytes) { uint8_t* r = p; p += (bytes + 255) & ~(size_t)255; return r; };
  float*  xc    = (float*) alloc((size_t)MROWS * CDIM * 4);        // 16 MB
  bf16_t* hbuf  = (bf16_t*)alloc((size_t)MROWS * CDIM * 2);        //  8 MB
  bf16_t* qkv   = (bf16_t*)alloc((size_t)MROWS * 3 * CDIM * 2);    // 24 MB
  float*  att   = (float*) alloc((size_t)MROWS * CDIM * 4);        // 16 MB
  bf16_t* hid   = (bf16_t*)alloc((size_t)MROWS * 4 * CDIM * 2);    // 32 MB
  bf16_t* wqkvb = (bf16_t*)alloc((size_t)3 * CDIM * CDIM * 2);     //  6 MB (also Wf)
  bf16_t* w1b   = (bf16_t*)alloc((size_t)4 * CDIM * CDIM * 2);     //  8 MB
  bf16_t* w2b   = (bf16_t*)alloc((size_t)4 * CDIM * CDIM * 2);     //  8 MB
  float*  seluc = (float*) alloc((size_t)BATCH * CDIM * 4);
  float*  m6    = (float*) alloc((size_t)LAYERS * BATCH * 6 * CDIM * 4);
  float*  m2    = (float*) alloc((size_t)BATCH * 2 * CDIM * 4);

  const int CC2 = CDIM * CDIM;

  (void)hipMemcpyAsync(xc, x, (size_t)MROWS * CDIM * 4, hipMemcpyDeviceToDevice, stream);

  cond_embed_kernel<<<(BATCH * CDIM) / 256, 256, 0, stream>>>(time_, label, Wl, bl, seluc);
  for (int l = 0; l < LAYERS; l++)
    cond_linear_kernel<<<6 * CDIM, 256, 0, stream>>>(
        seluc, Wm + (size_t)l * 6 * CC2, bm + (size_t)l * 6 * CDIM,
        m6 + (size_t)l * BATCH * 6 * CDIM, 6 * CDIM);
  cond_linear_kernel<<<2 * CDIM, 256, 0, stream>>>(seluc, Wmf, bmf, m2, 2 * CDIM);

  for (int l = 0; l < LAYERS; l++) {
    const float* m6l = m6 + (size_t)l * BATCH * 6 * CDIM;
    f2bf_kernel<<<CC2 / 256, 256, 0, stream>>>(Wq + (size_t)l * CC2, wqkvb, CC2);
    f2bf_kernel<<<CC2 / 256, 256, 0, stream>>>(Wk + (size_t)l * CC2, wqkvb + CC2, CC2);
    f2bf_kernel<<<CC2 / 256, 256, 0, stream>>>(Wv + (size_t)l * CC2, wqkvb + 2 * (size_t)CC2, CC2);
    f2bf_kernel<<<4 * CC2 / 256, 256, 0, stream>>>(W1 + (size_t)l * 4 * CC2, w1b, 4 * CC2);
    f2bf_kernel<<<4 * CC2 / 256, 256, 0, stream>>>(W2 + (size_t)l * 4 * CC2, w2b, 4 * CC2);

    ln_mod_kernel<<<MROWS, 256, 0, stream>>>(xc, m6l, 6 * CDIM, 0, CDIM, xmask, 1, hbuf);
    gemm_bf16_wmma_kernel<<<dim3(3 * CDIM / 128, MROWS / 128), 256, 0, stream>>>(
        hbuf, wqkvb, nullptr, nullptr, qkv, nullptr, nullptr, 0, nullptr,
        MROWS, 3 * CDIM, CDIM, 0);
    attn_wmma_kernel<<<dim3(BATCH * HEADS, NTOK / 64), 256, 0, stream>>>(qkv, xmask, att);
    resid_gate_kernel<<<(MROWS * CDIM) / 256, 256, 0, stream>>>(xc, att, m6l, 6 * CDIM, 2 * CDIM);
    ln_mod_kernel<<<MROWS, 256, 0, stream>>>(xc, m6l, 6 * CDIM, 3 * CDIM, 4 * CDIM, xmask, 0, hbuf);
    gemm_bf16_wmma_kernel<<<dim3(4 * CDIM / 128, MROWS / 128), 256, 0, stream>>>(
        hbuf, w1b, b1 + (size_t)l * 4 * CDIM, nullptr, hid, nullptr, nullptr, 0, nullptr,
        MROWS, 4 * CDIM, CDIM, 1);
    gemm_bf16_wmma_kernel<<<dim3(CDIM / 128, MROWS / 128), 256, 0, stream>>>(
        hid, w2b, b2 + (size_t)l * CDIM, nullptr, nullptr, xc, m6l + 5 * CDIM, 6 * CDIM, xmask,
        MROWS, CDIM, 4 * CDIM, 2);
  }

  f2bf_kernel<<<CC2 / 256, 256, 0, stream>>>(Wf, wqkvb, CC2);
  ln_mod_kernel<<<MROWS, 256, 0, stream>>>(xc, m2, 2 * CDIM, 0, CDIM, xmask, 0, hbuf);
  gemm_bf16_wmma_kernel<<<dim3(CDIM / 128, MROWS / 128), 256, 0, stream>>>(
      hbuf, wqkvb, bf_, out, nullptr, nullptr, nullptr, 0, xmask,
      MROWS, CDIM, CDIM, 3);

  (void)in_sizes; (void)n_in; (void)out_size; (void)ws_size;
}
